// SceneGenerator_59889023975663
// MI455X (gfx1250) — compile-verified, matrix operand order checked by reading, not run on hardware
//
#include <hip/hip_runtime.h>
#include <hip/hip_bf16.h>
#include <math.h>

#define CHN   128
#define LDIM  128
#define KMIX  8
#define NAC   6
#define BGR   8192
#define NN    (BGR*NAC)   // 49152

typedef __attribute__((ext_vector_type(16))) _Float16 v16h;
typedef __attribute__((ext_vector_type(8)))  _Float16 v8h;
typedef __attribute__((ext_vector_type(8)))  float    v8f;

// ---------------- threefry2x32 (JAX-style PRNG core) ----------------
__host__ __device__ inline void tf2x32(unsigned k0, unsigned k1,
                                       unsigned c0, unsigned c1,
                                       unsigned& o0, unsigned& o1) {
  unsigned ks0 = k0, ks1 = k1, ks2 = k0 ^ k1 ^ 0x1BD11BDAu;
  unsigned x0 = c0 + ks0, x1 = c1 + ks1;
  const int rotA[4] = {13, 15, 26, 6};
  const int rotB[4] = {17, 29, 16, 24};
  for (int i = 0; i < 5; ++i) {
    const int* rot = (i & 1) ? rotB : rotA;
    for (int r = 0; r < 4; ++r) {
      x0 += x1;
      x1 = (x1 << rot[r]) | (x1 >> (32 - rot[r]));
      x1 ^= x0;
    }
    unsigned inj = (unsigned)(i + 1);
    switch (i % 3) {
      case 0: x0 += ks1; x1 += ks2 + inj; break;
      case 1: x0 += ks2; x1 += ks0 + inj; break;
      default: x0 += ks0; x1 += ks1 + inj; break;
    }
  }
  o0 = x0; o1 = x1;
}
__device__ inline float u01(unsigned b) { return (b >> 8) * (1.0f / 16777216.0f); }

// ---------------- WMMA fragment helper ----------------
// 16-bit A/B fragment: lane (m = lane&15, h = lane>>4), two contiguous 8-half
// (16B) chunks at K offsets h*8 and 16+h*8 within each 32-wide K block.
__device__ inline v16h load_frag(const _Float16* p0, const _Float16* p1) {
  v8h lo = *(const v8h*)p0;
  v8h hi = *(const v8h*)p1;
  return __builtin_shufflevector(lo, hi, 0,1,2,3,4,5,6,7,8,9,10,11,12,13,14,15);
}

// ---------------- message-passing conv as WMMA GEMM ----------------
// out[i] = relu( x[i]@Wi + ((S_g - x[i])/5)@Wj + b )
// WT is [128][2*CIN] f16 (Wi^T | Wj^T concatenated along K).
template <int CIN>
__global__ __launch_bounds__(256) void conv_wmma(
    const _Float16* __restrict__ x,     // [NN][CIN]
    const _Float16* __restrict__ Sg,    // [BGR][CIN] graph sums
    const _Float16* __restrict__ WT,    // [128][2*CIN]
    const float*    __restrict__ bias,  // [128]
    _Float16*       __restrict__ y)     // [NN][128]
{
  constexpr int KT = 2 * CIN;
  __shared__ _Float16 As[16 * KT];
  const int tid = threadIdx.x;
  const int rowBase = blockIdx.x * 16;
  for (int idx = tid; idx < 16 * KT; idx += 256) {
    int r = idx / KT, k = idx - r * KT;
    int node = rowBase + r;
    float v;
    if (k < CIN) {
      v = (float)x[(size_t)node * CIN + k];
    } else {
      int kk = k - CIN;
      int g = node / NAC;
      v = ((float)Sg[(size_t)g * CIN + kk] - (float)x[(size_t)node * CIN + kk]) * 0.2f;
    }
    As[idx] = (_Float16)v;
  }
  __syncthreads();
  const int wave = tid >> 5, lane = tid & 31;
  const int m = lane & 15, h = lane >> 4;
  const int colBase = wave * 16;
  v8f acc = {};
  const _Float16* arow = &As[m * KT];
  const _Float16* brow = WT + (size_t)(colBase + m) * KT;
  for (int kb = 0; kb < KT; kb += 32) {
    v16h a = load_frag(arow + kb + h * 8, arow + kb + 16 + h * 8);
    v16h b = load_frag(brow + kb + h * 8, brow + kb + 16 + h * 8);
    acc = __builtin_amdgcn_wmma_f32_16x16x32_f16(false, a, false, b, (short)0, acc,
                                                 false, false);
  }
  float bv = bias[colBase + m];
#pragma unroll
  for (int v = 0; v < 8; ++v) {
    float t = acc[v] + bv;
    t = t > 0.f ? t : 0.f;
    int row = rowBase + h * 8 + v;
    y[(size_t)row * CHN + colBase + m] = (_Float16)t;
  }
}

// ---------------- generic WMMA GEMM: Y = act(X @ WT^T + b) ----------------
__global__ __launch_bounds__(256) void gemm_wmma(
    const _Float16* __restrict__ X,     // [M][K]
    const _Float16* __restrict__ WT,    // [Nout][K]
    const float*    __restrict__ bias,  // [NoutStore]
    _Float16*       __restrict__ outH,  // [M][Nout] or null
    float*          __restrict__ outF,  // [M][NoutStore] or null
    int K, int Nout, int NoutStore, int relu)
{
  __shared__ _Float16 As[16 * 768];
  const int tid = threadIdx.x;
  const int rowBase = blockIdx.x * 16;
  const int tot = 16 * K;
  for (int idx = tid; idx < tot; idx += 256) {
    int r = idx / K, k = idx - r * K;
    As[idx] = X[(size_t)(rowBase + r) * K + k];
  }
  __syncthreads();
  const int wave = tid >> 5, lane = tid & 31;
  const int m = lane & 15, h = lane >> 4;
  const int colBase = blockIdx.y * 128 + wave * 16;
  if (colBase >= Nout) return;  // wave-uniform: EXEC stays all-ones in live waves
  v8f acc = {};
  const _Float16* arow = &As[m * K];
  const _Float16* brow = WT + (size_t)(colBase + m) * K;
  for (int kb = 0; kb < K; kb += 32) {
    v16h a = load_frag(arow + kb + h * 8, arow + kb + 16 + h * 8);
    v16h b = load_frag(brow + kb + h * 8, brow + kb + 16 + h * 8);
    acc = __builtin_amdgcn_wmma_f32_16x16x32_f16(false, a, false, b, (short)0, acc,
                                                 false, false);
  }
  const int col = colBase + m;
  float bv = (col < NoutStore) ? bias[col] : 0.f;
#pragma unroll
  for (int v = 0; v < 8; ++v) {
    float t = acc[v] + bv;
    if (relu) t = t > 0.f ? t : 0.f;
    int row = rowBase + h * 8 + v;
    if (outH) outH[(size_t)row * Nout + col] = (_Float16)t;
    if (outF && col < NoutStore) outF[(size_t)row * NoutStore + col] = t;
  }
}

// ---------------- per-graph sums (and pooled means) ----------------
__global__ void graph_sum_k(const _Float16* __restrict__ act, int C,
                            _Float16* __restrict__ S,
                            _Float16* __restrict__ gdst, int gstride)
{
  int idx = blockIdx.x * blockDim.x + threadIdx.x;
  if (idx >= BGR * C) return;
  int g = idx / C, c = idx - g * C;
  float s = 0.f;
#pragma unroll
  for (int r = 0; r < NAC; ++r) s += (float)act[(size_t)(g * NAC + r) * C + c];
  S[(size_t)g * C + c] = (_Float16)s;
  if (gdst) gdst[(size_t)g * gstride + c] = (_Float16)(s * (1.0f / 6.0f));
}

// ---------------- input feature init: n0 [NN][384] (f16) ----------------
__global__ void init_nodes(
    const float* __restrict__ pos, const float* __restrict__ dimen,
    const float* __restrict__ vel, const float* __restrict__ atype,
    const float* __restrict__ lane, const float* __restrict__ dir,
    const float* __restrict__ w0, const float* __restrict__ b0,
    const float* __restrict__ w1, const float* __restrict__ b1,
    const float* __restrict__ w2, const float* __restrict__ b2,
    const float* __restrict__ w3, const float* __restrict__ b3,
    const float* __restrict__ w4, const float* __restrict__ b4,
    const float* __restrict__ w5, const float* __restrict__ b5,
    _Float16* __restrict__ act0)
{
  int idx = blockIdx.x * blockDim.x + threadIdx.x;
  if (idx >= NN * 384) return;
  int node = idx / 384, c = idx - node * 384;
  int seg = c >> 6, j = c & 63;
  float acc = 0.f;
  switch (seg) {
    case 0: acc = b0[j] + pos[node * 2] * w0[j] + pos[node * 2 + 1] * w0[64 + j]; break;
    case 1: acc = b1[j] + dimen[node * 2] * w1[j] + dimen[node * 2 + 1] * w1[64 + j]; break;
    case 2: acc = b2[j] + vel[node] * w2[j]; break;
    case 3: {
      acc = b3[j];
      for (int k = 0; k < 6; ++k) acc += atype[node * 6 + k] * w3[k * 64 + j];
      break;
    }
    case 4: acc = b4[j] + lane[node * 2] * w4[j] + lane[node * 2 + 1] * w4[64 + j]; break;
    default: {
      float dx = dir[node * 2], dy = dir[node * 2 + 1];
      float r = fmaxf(sqrtf(dx * dx + dy * dy), 1e-12f);
      acc = b5[j] + (dx / r) * w5[j] + (dy / r) * w5[64 + j];
      break;
    }
  }
  act0[idx] = (_Float16)(acc > 0.f ? acc : 0.f);
}

// ---------------- weight packing: f32 [din][dout] -> f16 WT[dout][ktot] ----
__global__ void pack_wt(const float* __restrict__ w, _Float16* __restrict__ wt,
                        int din, int dout, int ktot, int koff)
{
  int idx = blockIdx.x * blockDim.x + threadIdx.x;
  if (idx >= din * dout) return;
  int o = idx / din, k = idx - o * din;
  wt[(size_t)o * ktot + koff + k] = (_Float16)w[(size_t)k * dout + o];
}

// ---------------- softmax(pi) + hard Gumbel component selection ----------
__global__ void softmax_sample(const float* __restrict__ logits,
                               float* __restrict__ pi, int* __restrict__ ksel,
                               unsigned kg0, unsigned kg1)
{
  int b = blockIdx.x * blockDim.x + threadIdx.x;
  if (b >= BGR) return;
  float l[KMIX], mx = -1e30f;
  for (int k = 0; k < KMIX; ++k) { l[k] = logits[b * KMIX + k]; mx = fmaxf(mx, l[k]); }
  float s = 0.f, e[KMIX];
  for (int k = 0; k < KMIX; ++k) { e[k] = expf(l[k] - mx); s += e[k]; }
  for (int k = 0; k < KMIX; ++k) pi[b * KMIX + k] = e[k] / s;
  int best = 0; float bestv = -1e30f;
  for (int k = 0; k < KMIX; ++k) {
    unsigned w0, w1;
    tf2x32(kg0, kg1, 0u, (unsigned)(b * KMIX + k), w0, w1);
    float u = 1e-6f + u01(w0) * (1.0f - 2e-6f);
    float g = -logf(-logf(u));
    float t = l[k] + g;  // /0.5 temperature does not change argmax
    if (t > bestv) { bestv = t; best = k; }
  }
  ksel[b] = best;
}

// ---------------- z = mu[k*] + max(exp(.5 lv),1e-3)*eps  (f16) ------------
__global__ void make_z(const float* __restrict__ mu, const float* __restrict__ lv,
                       const int* __restrict__ ksel, _Float16* __restrict__ zh,
                       unsigned ke0, unsigned ke1)
{
  int idx = blockIdx.x * blockDim.x + threadIdx.x;
  if (idx >= BGR * LDIM) return;
  int b = idx / LDIM, d = idx - b * LDIM;
  int k = ksel[b];
  size_t o = (size_t)b * (KMIX * LDIM) + (size_t)k * LDIM + d;
  float m = mu[o], v = lv[o];
  float sd = fmaxf(expf(0.5f * v), 1e-3f);
  unsigned w0, w1;
  tf2x32(ke0, ke1, 0u, (unsigned)((b * KMIX + k) * LDIM + d), w0, w1);
  float u1 = fmaxf(u01(w0), 1e-7f);
  float u2 = u01(w1);
  float eps = sqrtf(-2.f * logf(u1)) * cosf(6.28318530717958647692f * u2);
  zh[idx] = (_Float16)(m + sd * eps);
}

// ---------------- tiny head output linear: [NN][128] -> [NN][dout] --------
__global__ void head_out(const _Float16* __restrict__ hid,
                         const float* __restrict__ w, const float* __restrict__ b,
                         float* __restrict__ out, int dout, int stride)
{
  int idx = blockIdx.x * blockDim.x + threadIdx.x;
  if (idx >= NN * dout) return;
  int n = idx / dout, j = idx - n * dout;
  float s = b[j];
  for (int k = 0; k < CHN; ++k) s += (float)hid[(size_t)n * CHN + k] * w[k * dout + j];
  out[(size_t)n * stride + j] = s;
}

__global__ void normalize_rows2(float* __restrict__ d)
{
  int n = blockIdx.x * blockDim.x + threadIdx.x;
  if (n >= NN) return;
  float x = d[2 * n], y = d[2 * n + 1];
  float r = fmaxf(sqrtf(x * x + y * y), 1e-12f);
  d[2 * n] = x / r;
  d[2 * n + 1] = y / r;
}

// =========================================================================
static inline char* wsalloc(void* base, size_t& off, size_t bytes) {
  char* p = (char*)base + off;
  off += (bytes + 255) & ~(size_t)255;
  return p;
}

extern "C" void kernel_launch(void* const* d_in, const int* in_sizes, int n_in,
                              void* d_out, int out_size, void* d_ws, size_t ws_size,
                              hipStream_t stream) {
  (void)in_sizes; (void)n_in; (void)out_size; (void)ws_size;
  // ---- inputs (insertion order, params recursively flattened) ----
  const float* pos    = (const float*)d_in[0];
  const float* dimen  = (const float*)d_in[1];
  const float* vel    = (const float*)d_in[2];
  const float* atype  = (const float*)d_in[3];
  const float* lane   = (const float*)d_in[4];
  const float* dir    = (const float*)d_in[5];
  // d_in[6] edge_index, d_in[7] batch: structure exploited analytically.
  const int P = 8;  // params start
  // init linears: pos(8,9) size(10,11) vel(12,13) type(14,15) dir(16,17) lane(18,19)
  // convs: e1(20..22) e2(23..25) e3(26..28) d1(29..31) d2(32..34) d3(35..37)
  // d_ft_init(38,39) aggregate(40,41) fc_mu(42,43) fc_var(44,45) fc_pi(46,47)
  // heads: 48 + 4*i : {w0,b0,w1,b1}
  float* out = (float*)d_out;
  float* o_pos  = out;
  float* o_size = out + 2 * NN;
  float* o_vel  = out + 4 * NN;
  float* o_act  = out + 5 * NN;
  float* o_dir  = out + 11 * NN;
  float* o_lane = out + 13 * NN;
  float* o_lv   = out + 15 * NN;
  float* o_mu   = o_lv + (size_t)BGR * 1024;
  float* o_pi   = o_mu + (size_t)BGR * 1024;
  float* o_pl   = o_pi + (size_t)BGR * 8;

  // ---- workspace layout ----
  size_t off = 0;
  _Float16* WT_e1  = (_Float16*)wsalloc(d_ws, off, (size_t)128 * 768 * 2);
  _Float16* WT_e2  = (_Float16*)wsalloc(d_ws, off, (size_t)128 * 256 * 2);
  _Float16* WT_e3  = (_Float16*)wsalloc(d_ws, off, (size_t)128 * 256 * 2);
  _Float16* WT_d1  = (_Float16*)wsalloc(d_ws, off, (size_t)128 * 256 * 2);
  _Float16* WT_d2  = (_Float16*)wsalloc(d_ws, off, (size_t)128 * 256 * 2);
  _Float16* WT_d3  = (_Float16*)wsalloc(d_ws, off, (size_t)128 * 256 * 2);
  _Float16* WT_agg = (_Float16*)wsalloc(d_ws, off, (size_t)128 * 768 * 2);
  _Float16* WT_mu  = (_Float16*)wsalloc(d_ws, off, (size_t)1024 * 128 * 2);
  _Float16* WT_var = (_Float16*)wsalloc(d_ws, off, (size_t)1024 * 128 * 2);
  _Float16* WT_pi  = (_Float16*)wsalloc(d_ws, off, (size_t)16 * 128 * 2);
  _Float16* WT_dft = (_Float16*)wsalloc(d_ws, off, (size_t)768 * 128 * 2);
  _Float16* WT_hd  = (_Float16*)wsalloc(d_ws, off, (size_t)9 * 128 * 128 * 2);
  _Float16* act0   = (_Float16*)wsalloc(d_ws, off, (size_t)NN * 384 * 2);
  _Float16* Sbuf   = (_Float16*)wsalloc(d_ws, off, (size_t)BGR * 384 * 2);
  _Float16* gfeat  = (_Float16*)wsalloc(d_ws, off, (size_t)BGR * 768 * 2);
  _Float16* actA   = (_Float16*)wsalloc(d_ws, off, (size_t)NN * 128 * 2);
  _Float16* actB   = (_Float16*)wsalloc(d_ws, off, (size_t)NN * 128 * 2);
  _Float16* latH   = (_Float16*)wsalloc(d_ws, off, (size_t)BGR * 128 * 2);
  _Float16* zH     = (_Float16*)wsalloc(d_ws, off, (size_t)BGR * 128 * 2);
  int*      ksel   = (int*)wsalloc(d_ws, off, (size_t)BGR * 4);

  // ---- PRNG child keys from root key (0, 42) ----
  unsigned kg0, kg1, ke0, ke1;
  tf2x32(0u, 42u, 0u, 1u, kg0, kg1);
  tf2x32(0u, 42u, 2u, 3u, ke0, ke1);

  const int T = 256;
  auto blks = [](long long n, int t) { return (unsigned)((n + t - 1) / t); };
  #define PACK(widx, wt, din, dout, ktot, koff) \
    pack_wt<<<blks((long long)(din) * (dout), T), T, 0, stream>>>( \
        (const float*)d_in[widx], wt, din, dout, ktot, koff)

  // ---- pack all GEMM weights to f16 transposed layout ----
  PACK(20, WT_e1, 384, 128, 768, 0);   PACK(21, WT_e1, 384, 128, 768, 384);
  PACK(23, WT_e2, 128, 128, 256, 0);   PACK(24, WT_e2, 128, 128, 256, 128);
  PACK(26, WT_e3, 128, 128, 256, 0);   PACK(27, WT_e3, 128, 128, 256, 128);
  PACK(29, WT_d1, 128, 128, 256, 0);   PACK(30, WT_d1, 128, 128, 256, 128);
  PACK(32, WT_d2, 128, 128, 256, 0);   PACK(33, WT_d2, 128, 128, 256, 128);
  PACK(35, WT_d3, 128, 128, 256, 0);   PACK(36, WT_d3, 128, 128, 256, 128);
  PACK(40, WT_agg, 768, 128, 768, 0);
  PACK(42, WT_mu, 128, 1024, 128, 0);
  PACK(44, WT_var, 128, 1024, 128, 0);
  hipMemsetAsync(WT_pi, 0, (size_t)16 * 128 * 2, stream);
  PACK(46, WT_pi, 128, 8, 128, 0);
  PACK(38, WT_dft, 128, 768, 128, 0);
  for (int i = 0; i < 9; ++i)
    PACK(48 + 4 * i, WT_hd + (size_t)i * 128 * 128, 128, 128, 128, 0);

  // ---- encoder ----
  init_nodes<<<blks((long long)NN * 384, T), T, 0, stream>>>(
      pos, dimen, vel, atype, lane, dir,
      (const float*)d_in[P + 0], (const float*)d_in[P + 1],
      (const float*)d_in[P + 2], (const float*)d_in[P + 3],
      (const float*)d_in[P + 4], (const float*)d_in[P + 5],
      (const float*)d_in[P + 6], (const float*)d_in[P + 7],
      (const float*)d_in[18],    (const float*)d_in[19],   // lane_index_init
      (const float*)d_in[16],    (const float*)d_in[17],   // direction_init
      act0);
  graph_sum_k<<<blks((long long)BGR * 384, T), T, 0, stream>>>(act0, 384, Sbuf, gfeat + 0, 768);
  conv_wmma<384><<<NN / 16, T, 0, stream>>>(act0, Sbuf, WT_e1, (const float*)d_in[22], actA);
  graph_sum_k<<<blks((long long)BGR * 128, T), T, 0, stream>>>(actA, 128, Sbuf, gfeat + 384, 768);
  conv_wmma<128><<<NN / 16, T, 0, stream>>>(actA, Sbuf, WT_e2, (const float*)d_in[25], actB);
  graph_sum_k<<<blks((long long)BGR * 128, T), T, 0, stream>>>(actB, 128, Sbuf, gfeat + 512, 768);
  conv_wmma<128><<<NN / 16, T, 0, stream>>>(actB, Sbuf, WT_e3, (const float*)d_in[28], actA);
  graph_sum_k<<<blks((long long)BGR * 128, T), T, 0, stream>>>(actA, 128, Sbuf, gfeat + 640, 768);

  // ---- latent heads ----
  gemm_wmma<<<dim3(BGR / 16, 1), T, 0, stream>>>(gfeat, WT_agg, (const float*)d_in[41],
                                                 latH, nullptr, 768, 128, 128, 0);
  gemm_wmma<<<dim3(BGR / 16, 8), T, 0, stream>>>(latH, WT_mu, (const float*)d_in[43],
                                                 nullptr, o_mu, 128, 1024, 1024, 0);
  gemm_wmma<<<dim3(BGR / 16, 8), T, 0, stream>>>(latH, WT_var, (const float*)d_in[45],
                                                 nullptr, o_lv, 128, 1024, 1024, 0);
  gemm_wmma<<<dim3(BGR / 16, 1), T, 0, stream>>>(latH, WT_pi, (const float*)d_in[47],
                                                 nullptr, o_pl, 128, 16, 8, 0);
  softmax_sample<<<blks(BGR, T), T, 0, stream>>>(o_pl, o_pi, ksel, kg0, kg1);
  make_z<<<blks((long long)BGR * LDIM, T), T, 0, stream>>>(o_mu, o_lv, ksel, zH, ke0, ke1);

  // ---- decoder ----
  gemm_wmma<<<dim3(BGR / 16, 6), T, 0, stream>>>(zH, WT_dft, (const float*)d_in[39],
                                                 actA, nullptr, 128, 768, 768, 1);
  graph_sum_k<<<blks((long long)BGR * 128, T), T, 0, stream>>>(actA, 128, Sbuf, nullptr, 0);
  conv_wmma<128><<<NN / 16, T, 0, stream>>>(actA, Sbuf, WT_d1, (const float*)d_in[31], actB);
  graph_sum_k<<<blks((long long)BGR * 128, T), T, 0, stream>>>(actB, 128, Sbuf, nullptr, 0);
  conv_wmma<128><<<NN / 16, T, 0, stream>>>(actB, Sbuf, WT_d2, (const float*)d_in[34], actA);
  graph_sum_k<<<blks((long long)BGR * 128, T), T, 0, stream>>>(actA, 128, Sbuf, nullptr, 0);
  conv_wmma<128><<<NN / 16, T, 0, stream>>>(actA, Sbuf, WT_d3, (const float*)d_in[37], actB);
  // d3 lives in actB

  // ---- output heads ----
  struct HeadTgt { float* base; int dout; int stride; };
  const HeadTgt tgt[9] = {
      {o_pos + 0, 1, 2}, {o_pos + 1, 1, 2}, {o_size + 0, 1, 2}, {o_size + 1, 1, 2},
      {o_vel, 1, 1},     {o_act, 6, 6},     {o_dir + 0, 1, 2},  {o_dir + 1, 1, 2},
      {o_lane, 2, 2}};
  for (int i = 0; i < 9; ++i) {
    gemm_wmma<<<dim3(NN / 16, 1), T, 0, stream>>>(
        actB, WT_hd + (size_t)i * 128 * 128, (const float*)d_in[49 + 4 * i],
        actA, nullptr, 128, 128, 128, 1);
    head_out<<<blks((long long)NN * tgt[i].dout, T), T, 0, stream>>>(
        actA, (const float*)d_in[50 + 4 * i], (const float*)d_in[51 + 4 * i],
        tgt[i].base, tgt[i].dout, tgt[i].stride);
  }
  normalize_rows2<<<blks(NN, T), T, 0, stream>>>(o_dir);
}
